// RBM_8443905704555
// MI455X (gfx1250) — matrix operand order repaired and checked by reading, not compiled, not run on hardware
//
#include <hip/hip_runtime.h>

typedef __attribute__((ext_vector_type(16))) __bf16 v16bf;
typedef __attribute__((ext_vector_type(8)))  float  v8f;

#define NVIS 2048   // K
#define NHID 512    // N
#define MTILE 64    // batch rows per workgroup (4 WMMA M-tiles per wave)

__device__ __forceinline__ float softplus_f(float x) {
    // stable: max(x,0) + log1p(exp(-|x|)); fast exp/log -> v_exp_f32 / v_log_f32
    return fmaxf(x, 0.0f) + __logf(1.0f + __expf(-fabsf(x)));
}

__device__ __forceinline__ v16bf cvt_frag(v8f lo, v8f hi) {
    v16bf f;
    #pragma unroll
    for (int i = 0; i < 8; ++i) {
        f[i]     = (__bf16)lo[i];
        f[i + 8] = (__bf16)hi[i];
    }
    return f;
}

// ---- prep: W f32 -> bf16 (row-major, same layout) -------------------------
__global__ void w_to_bf16_kernel(const float* __restrict__ W,
                                 __bf16* __restrict__ Wb, int n) {
    int i = (blockIdx.x * blockDim.x + threadIdx.x) * 4;
    if (i + 3 < n) {
        float4 f = *(const float4*)(W + i);
        Wb[i + 0] = (__bf16)f.x;
        Wb[i + 1] = (__bf16)f.y;
        Wb[i + 2] = (__bf16)f.z;
        Wb[i + 3] = (__bf16)f.w;
    }
}

// ---- fused: GEMM (bf16 WMMA, f32 acc) + softplus-reduce + softmax ---------
__global__ __launch_bounds__(256, 1)
void rbm_fused_kernel(const float*  __restrict__ A,   // [B, 2048] f32
                      const __bf16* __restrict__ Wb,  // [512, 2048] bf16
                      const float*  __restrict__ U,   // [512, 2]
                      const float*  __restrict__ hb,  // [512]
                      const float*  __restrict__ yb,  // [2]
                      float*        __restrict__ out) // [B, 2]
{
    __shared__ float red[8][MTILE][2];

    const int lane = threadIdx.x & 31;
    const int wave = threadIdx.x >> 5;          // 0..7, each owns 4 N-tiles
    const int half = lane >> 4;                 // 0: lanes 0-15, 1: lanes 16-31
    const int l16  = lane & 15;
    const int row0 = blockIdx.x * MTILE;        // batch rows [row0, row0+64)

    // A fragment bases (4 M-tiles): contiguous K order matching B:
    //   lanes 0-15 : row=l16, K = kb+{0..15}
    //   lanes 16-31: row=l16, K = kb+{16..31}
    const float* arow0 = A + (size_t)(row0 + l16) * NVIS + (half ? 16 : 0);
    const float* arow1 = arow0 + (size_t)16 * NVIS;
    const float* arow2 = arow0 + (size_t)32 * NVIS;
    const float* arow3 = arow0 + (size_t)48 * NVIS;

    // B fragment bases (4 N-tiles): lane = column n, K = kb+{0..15 | 16..31}
    const __bf16* brow0 = Wb + (size_t)((wave * 4 + 0) * 16 + l16) * NVIS + (half ? 16 : 0);
    const __bf16* brow1 = Wb + (size_t)((wave * 4 + 1) * 16 + l16) * NVIS + (half ? 16 : 0);
    const __bf16* brow2 = Wb + (size_t)((wave * 4 + 2) * 16 + l16) * NVIS + (half ? 16 : 0);
    const __bf16* brow3 = Wb + (size_t)((wave * 4 + 3) * 16 + l16) * NVIS + (half ? 16 : 0);

    v8f acc[4][4];
    #pragma unroll
    for (int m = 0; m < 4; ++m)
        #pragma unroll
        for (int t = 0; t < 4; ++t)
            acc[m][t] = (v8f){};

    for (int kb = 0; kb < NVIS; kb += 32) {
        // prefetch the A streams ~8 K-steps ahead (speculative, no LOADcnt)
        __builtin_prefetch(arow0 + kb + 256, 0, 0);
        __builtin_prefetch(arow1 + kb + 256, 0, 0);
        __builtin_prefetch(arow2 + kb + 256, 0, 0);
        __builtin_prefetch(arow3 + kb + 256, 0, 0);

        // B fragments for this K-step (reused by all 4 M-groups)
        v16bf b0 = *(const v16bf*)(brow0 + kb);
        v16bf b1 = *(const v16bf*)(brow1 + kb);
        v16bf b2 = *(const v16bf*)(brow2 + kb);
        v16bf b3 = *(const v16bf*)(brow3 + kb);

        // ---- software-pipelined A: ping-pong fragments afA/afB ----
        v8f lo = *(const v8f*)(arow0 + kb);
        v8f hi = *(const v8f*)(arow0 + kb + 8);
        v16bf afA = cvt_frag(lo, hi);

        // m = 0 (consume afA, stage m=1 into afB)
        lo = *(const v8f*)(arow1 + kb);
        hi = *(const v8f*)(arow1 + kb + 8);
        acc[0][0] = __builtin_amdgcn_wmma_f32_16x16x32_bf16(false, afA, false, b0, (short)0, acc[0][0], false, false);
        acc[0][1] = __builtin_amdgcn_wmma_f32_16x16x32_bf16(false, afA, false, b1, (short)0, acc[0][1], false, false);
        acc[0][2] = __builtin_amdgcn_wmma_f32_16x16x32_bf16(false, afA, false, b2, (short)0, acc[0][2], false, false);
        acc[0][3] = __builtin_amdgcn_wmma_f32_16x16x32_bf16(false, afA, false, b3, (short)0, acc[0][3], false, false);
        v16bf afB = cvt_frag(lo, hi);

        // m = 1 (consume afB, stage m=2 into afA)
        lo = *(const v8f*)(arow2 + kb);
        hi = *(const v8f*)(arow2 + kb + 8);
        acc[1][0] = __builtin_amdgcn_wmma_f32_16x16x32_bf16(false, afB, false, b0, (short)0, acc[1][0], false, false);
        acc[1][1] = __builtin_amdgcn_wmma_f32_16x16x32_bf16(false, afB, false, b1, (short)0, acc[1][1], false, false);
        acc[1][2] = __builtin_amdgcn_wmma_f32_16x16x32_bf16(false, afB, false, b2, (short)0, acc[1][2], false, false);
        acc[1][3] = __builtin_amdgcn_wmma_f32_16x16x32_bf16(false, afB, false, b3, (short)0, acc[1][3], false, false);
        afA = cvt_frag(lo, hi);

        // m = 2 (consume afA, stage m=3 into afB)
        lo = *(const v8f*)(arow3 + kb);
        hi = *(const v8f*)(arow3 + kb + 8);
        acc[2][0] = __builtin_amdgcn_wmma_f32_16x16x32_bf16(false, afA, false, b0, (short)0, acc[2][0], false, false);
        acc[2][1] = __builtin_amdgcn_wmma_f32_16x16x32_bf16(false, afA, false, b1, (short)0, acc[2][1], false, false);
        acc[2][2] = __builtin_amdgcn_wmma_f32_16x16x32_bf16(false, afA, false, b2, (short)0, acc[2][2], false, false);
        acc[2][3] = __builtin_amdgcn_wmma_f32_16x16x32_bf16(false, afA, false, b3, (short)0, acc[2][3], false, false);
        afB = cvt_frag(lo, hi);

        // m = 3 (consume afB)
        acc[3][0] = __builtin_amdgcn_wmma_f32_16x16x32_bf16(false, afB, false, b0, (short)0, acc[3][0], false, false);
        acc[3][1] = __builtin_amdgcn_wmma_f32_16x16x32_bf16(false, afB, false, b1, (short)0, acc[3][1], false, false);
        acc[3][2] = __builtin_amdgcn_wmma_f32_16x16x32_bf16(false, afB, false, b2, (short)0, acc[3][2], false, false);
        acc[3][3] = __builtin_amdgcn_wmma_f32_16x16x32_bf16(false, afB, false, b3, (short)0, acc[3][3], false, false);
    }

    // Epilogue: pf = acc + h_bias[n]; accumulate softplus(pf + U[n,c]) over n.
    // C/D layout: slot v of lane -> element (row = v + 8*half, col = l16).
    float p0[4][8], p1[4][8];
    #pragma unroll
    for (int m = 0; m < 4; ++m)
        #pragma unroll
        for (int v = 0; v < 8; ++v) { p0[m][v] = 0.0f; p1[m][v] = 0.0f; }

    #pragma unroll
    for (int t = 0; t < 4; ++t) {
        const int n = (wave * 4 + t) * 16 + l16;
        const float h  = hb[n];
        const float u0 = U[2 * n + 0];
        const float u1 = U[2 * n + 1];
        #pragma unroll
        for (int m = 0; m < 4; ++m) {
            #pragma unroll
            for (int v = 0; v < 8; ++v) {
                const float pf = acc[m][t][v] + h;
                p0[m][v] += softplus_f(pf + u0);
                p1[m][v] += softplus_f(pf + u1);
            }
        }
    }

    // reduce across the 16 columns held by each half-wave (xor masks < 16
    // keep the reduction within lanes 0-15 / 16-31 independently)
    #pragma unroll
    for (int m = 0; m < 4; ++m) {
        #pragma unroll
        for (int v = 0; v < 8; ++v) {
            #pragma unroll
            for (int s = 1; s < 16; s <<= 1) {
                p0[m][v] += __shfl_xor(p0[m][v], s, 32);
                p1[m][v] += __shfl_xor(p1[m][v], s, 32);
            }
        }
    }

    if (l16 == 0) {
        #pragma unroll
        for (int m = 0; m < 4; ++m) {
            #pragma unroll
            for (int v = 0; v < 8; ++v) {
                red[wave][m * 16 + half * 8 + v][0] = p0[m][v];
                red[wave][m * 16 + half * 8 + v][1] = p1[m][v];
            }
        }
    }
    __syncthreads();

    // final reduce over 8 waves + 2-class softmax, one thread per batch row
    if (threadIdx.x < MTILE) {
        const int r = threadIdx.x;
        float l0 = yb[0], l1 = yb[1];
        #pragma unroll
        for (int w = 0; w < 8; ++w) {
            l0 += red[w][r][0];
            l1 += red[w][r][1];
        }
        const float mx = fmaxf(l0, l1);
        const float e0 = __expf(l0 - mx);
        const float e1 = __expf(l1 - mx);
        const float inv = 1.0f / (e0 + e1);
        out[(size_t)(row0 + r) * 2 + 0] = e0 * inv;
        out[(size_t)(row0 + r) * 2 + 1] = e1 * inv;
    }
}

extern "C" void kernel_launch(void* const* d_in, const int* in_sizes, int n_in,
                              void* d_out, int out_size, void* d_ws, size_t ws_size,
                              hipStream_t stream) {
    const float* A  = (const float*)d_in[0];   // input_data [16384, 2048]
    const float* W  = (const float*)d_in[1];   // W [512, 2048]
    const float* U  = (const float*)d_in[2];   // U [512, 2]
    const float* hb = (const float*)d_in[3];   // h_bias [512]
    const float* yb = (const float*)d_in[4];   // y_bias [2]
    float* out = (float*)d_out;                // [16384, 2]

    __bf16* Wb = (__bf16*)d_ws;                // 512*2048*2 = 2 MB scratch

    const int nW = in_sizes[1];                // 1048576
    w_to_bf16_kernel<<<(nW / 4 + 255) / 256, 256, 0, stream>>>(W, Wb, nW);

    const int B = in_sizes[0] / NVIS;          // 16384
    rbm_fused_kernel<<<B / MTILE, 256, 0, stream>>>(A, Wb, U, hb, yb, out);
}
